// GCNLayerNorm_84954453115108
// MI455X (gfx1250) — compile-verified, matrix operand order checked by reading, not run on hardware
//
#include <hip/hip_runtime.h>
#include <hip/hip_bf16.h>

// ---------------------------------------------------------------------------
// GCN layer: h = LN( norm * Agg( (xW)*norm ) + xWr + br ) -> ReLU
// N=50000, E=800000, DIM_IN=384, DIM_OUT=512
//  - f16 WMMA (v_wmma_f32_16x16x32_f16) fused GEMM for xW and xWr (shared A).
//    Fully-unrolled, software-pipelined K-loop: ping/pong fragment sets so
//    b128 fragment loads overlap the previous step's 4 WMMAs.
//  - CSR-by-dst build on device; gather-based aggregation (no fp atomics in
//    the hot loop; hs (102MB) is L2-resident on MI455X's 192MB L2).
//  - Fused norm/residual/LayerNorm/ReLU epilogue.
// ---------------------------------------------------------------------------

typedef __attribute__((ext_vector_type(16))) _Float16 v16h;
typedef __attribute__((ext_vector_type(8)))  float    v8f;

#define DIM_IN   384
#define DIM_OUT  512
#define NPRIME   1024          // fused N dimension: [0,512)=W, [512,1024)=Wr
#define KPAIRS   (DIM_IN / 2)  // 192
#define KSTEPS   (DIM_IN / 32) // 12

union F16Frag {
    unsigned int u[8];
    uint4        q[2];
    v16h         h;
};

static __device__ __forceinline__ unsigned int pack2f16(float a, float b) {
    union { _Float16 h[2]; unsigned int u; } p;
    p.h[0] = (_Float16)a; p.h[1] = (_Float16)b;
    return p.u;
}

// ---------------- 1. zero int scratch --------------------------------------
__global__ void zero_kernel(int* p, int n) {
    for (int i = blockIdx.x * blockDim.x + threadIdx.x; i < n;
         i += gridDim.x * blockDim.x)
        p[i] = 0;
}

// ---------------- 2. in-degree count ---------------------------------------
__global__ void count_kernel(const int* __restrict__ dst, int* __restrict__ cnt,
                             int nEdges) {
    int e = blockIdx.x * blockDim.x + threadIdx.x;
    if (e < nEdges) atomicAdd(&cnt[dst[e]], 1);
}

// ---------------- 3. single-block exclusive scan + norm --------------------
__global__ void scan_kernel(const int* __restrict__ cnt, int* __restrict__ off,
                            float* __restrict__ normv, int n) {
    __shared__ int sh[1024];
    const int tid = threadIdx.x;
    int carry = 0;
    for (int base = 0; base < n; base += 1024) {
        int idx = base + tid;
        int v = (idx < n) ? cnt[idx] : 0;
        sh[tid] = v;
        __syncthreads();
        #pragma unroll
        for (int s = 1; s < 1024; s <<= 1) {
            int t = (tid >= s) ? sh[tid - s] : 0;
            __syncthreads();
            sh[tid] += t;
            __syncthreads();
        }
        if (idx < n) {
            off[idx]   = carry + sh[tid] - v;           // exclusive
            normv[idx] = rsqrtf((float)(v + 1));        // +1 self loop, >=1
        }
        carry += sh[1023];
        __syncthreads();
    }
    if (tid == 0) off[n] = carry;
}

// ---------------- 4. CSR fill (src ids bucketed by dst) --------------------
__global__ void fill_kernel(const int* __restrict__ src,
                            const int* __restrict__ dst,
                            const int* __restrict__ off,
                            int* __restrict__ cursor,
                            int* __restrict__ elist, int nEdges) {
    int e = blockIdx.x * blockDim.x + threadIdx.x;
    if (e < nEdges) {
        int d = dst[e];
        int pos = atomicAdd(&cursor[d], 1);
        elist[off[d] + pos] = src[e];
    }
}

// ---------------- 5. x (f32) -> xh (f16 pairs, row-major) ------------------
__global__ void convx_kernel(const float* __restrict__ x,
                             unsigned int* __restrict__ xh, int nPairs) {
    int i = blockIdx.x * blockDim.x + threadIdx.x;
    if (i < nPairs) xh[i] = pack2f16(x[2 * i], x[2 * i + 1]);
}

// ------- 6. [W|Wr] (f32) -> Wp (f16, B-fragment-contiguous layout) ---------
// Wp[(kk*1024 + c)*16 + kl] packs rows k=32*kk+2*kl, k+1 of column c of the
// column-concatenation W' = [W | Wr] (384 x 1024). A lane (col c, half h)
// then reads its 8 fragment words as two contiguous uint4:
//   words [h*4 .. h*4+3] and [h*4+8 .. h*4+11] of the 16-word group.
__global__ void convw_kernel(const float* __restrict__ W,
                             const float* __restrict__ Wr,
                             unsigned int* __restrict__ Wp, int total) {
    int i = blockIdx.x * blockDim.x + threadIdx.x;
    if (i >= total) return;
    const int kk  = i >> 14;                 // / (1024*16)
    const int rem = i & 16383;
    const int c   = rem >> 4;
    const int kl  = rem & 15;
    const int k   = kk * 32 + kl * 2;
    float a, b;
    if (c < DIM_OUT) {
        a = W[(size_t)k       * DIM_OUT + c];
        b = W[(size_t)(k + 1) * DIM_OUT + c];
    } else {
        const int cc = c - DIM_OUT;
        a = Wr[(size_t)k       * DIM_OUT + cc];
        b = Wr[(size_t)(k + 1) * DIM_OUT + cc];
    }
    Wp[i] = pack2f16(a, b);
}

// ---------------- 7. fused WMMA GEMM ---------------------------------------
// Block = 256 threads = 8 waves, tile 64(M) x 128(N'). Wave (wm,wn) computes a
// 16x64 strip: 4 x v_wmma_f32_16x16x32_f16 per K-step, A-frag reused 4x.
// blockIdx.y < 4  -> cols [0,512):   hs[r][c] = dot * norm[r]
// blockIdx.y >= 4 -> cols [512,1024): d_out[r][c-512] = dot + br[c-512]

static __device__ __forceinline__ void load_frags(const uint4* __restrict__ ap,
                                                  const uint4* __restrict__ bp,
                                                  int kk, F16Frag& A,
                                                  F16Frag B[4]) {
    A.q[0] = ap[kk * 4 + 0];
    A.q[1] = ap[kk * 4 + 2];
    #pragma unroll
    for (int t = 0; t < 4; ++t) {
        B[t].q[0] = bp[kk * 4096 + t * 64 + 0];
        B[t].q[1] = bp[kk * 4096 + t * 64 + 2];
    }
}

static __device__ __forceinline__ void do_wmma4(v8f acc[4], const F16Frag& A,
                                                const F16Frag B[4]) {
    #pragma unroll
    for (int t = 0; t < 4; ++t)
        acc[t] = __builtin_amdgcn_wmma_f32_16x16x32_f16(
            false, A.h, false, B[t].h, (short)0, acc[t], false, false);
}

__global__ __launch_bounds__(256)
void gemm_kernel(const unsigned int* __restrict__ xh,   // nNodes x 192
                 const unsigned int* __restrict__ Wp,   // (12*1024*16) words
                 const float* __restrict__ normv,
                 const float* __restrict__ br,
                 float* __restrict__ hs,                // nNodes x 512
                 float* __restrict__ res_out,           // nNodes x 512 (d_out)
                 int nNodes) {
    const int lane = threadIdx.x & 31;
    const int wave = threadIdx.x >> 5;
    const int wm   = wave & 3;            // 0..3 -> M sub-tile
    const int wn   = wave >> 2;           // 0..1 -> N strip of 64
    const int half = lane >> 4;           // 0: lanes 0-15, 1: lanes 16-31
    const int l16  = lane & 15;

    const int blockM = blockIdx.x * 64;
    const int blockN = blockIdx.y * 128;

    const int row  = blockM + wm * 16 + l16;       // A row this lane supplies
    const int arow = (row < nNodes) ? row : (nNodes - 1);
    const int col0 = blockN + wn * 64 + l16;       // base B/D column

    // Base pointers; ALL K-loop loads are immediate offsets from these.
    // A uint4 index: kk*4 + {0,2}           (row has 48 uint4)
    // B uint4 index: kk*4096 + t*64 + {0,2} (max byte off ~733KB < 2^23)
    const uint4* ap = (const uint4*)(xh + (size_t)arow * KPAIRS) + half;
    const uint4* bp = (const uint4*)Wp + ((size_t)col0 << 2) + half;

    v8f acc[4] = {v8f{}, v8f{}, v8f{}, v8f{}};

    // 2-stage software pipeline: loads for step kk+1 are issued before the
    // WMMAs of step kk, so fragment loads overlap matrix math.
    F16Frag Aa, Ab;
    F16Frag Ba[4], Bb[4];
    load_frags(ap, bp, 0, Aa, Ba);
    #pragma unroll
    for (int kk = 0; kk < KSTEPS; kk += 2) {
        load_frags(ap, bp, kk + 1, Ab, Bb);
        do_wmma4(acc, Aa, Ba);
        if (kk + 2 < KSTEPS) load_frags(ap, bp, kk + 2, Aa, Ba);
        do_wmma4(acc, Ab, Bb);
    }

    // Epilogue. D layout: VGPR v, lanes 0-15 -> M=v, lanes 16-31 -> M=v+8.
    const int rbase = blockM + wm * 16 + half * 8;
    if (blockN < DIM_OUT) {                    // scalar-uniform branch
        float nm[8];
        #pragma unroll
        for (int v = 0; v < 8; ++v) {
            const int r = rbase + v;
            nm[v] = (r < nNodes) ? normv[r] : 0.0f;
        }
        #pragma unroll
        for (int t = 0; t < 4; ++t) {
            const int c = col0 + t * 16;
            #pragma unroll
            for (int v = 0; v < 8; ++v) {
                const int r = rbase + v;
                if (r < nNodes)
                    hs[(size_t)r * DIM_OUT + c] = acc[t][v] * nm[v];
            }
        }
    } else {
        float brv[4];
        #pragma unroll
        for (int t = 0; t < 4; ++t)
            brv[t] = br[col0 + t * 16 - DIM_OUT];
        #pragma unroll
        for (int t = 0; t < 4; ++t) {
            const int c = col0 + t * 16 - DIM_OUT;
            #pragma unroll
            for (int v = 0; v < 8; ++v) {
                const int r = rbase + v;
                if (r < nNodes)
                    res_out[(size_t)r * DIM_OUT + c] = acc[t][v] + brv[t];
            }
        }
    }
}

// ---------------- 8. gather-aggregate + norm + residual + LN + ReLU --------
// One block (128 threads, 4 waves) per node; thread owns 4 consecutive cols.
__global__ __launch_bounds__(128)
void agg_kernel(const float* __restrict__ hs,
                const int* __restrict__ off,
                const int* __restrict__ elist,
                const float* __restrict__ normv,
                const float* __restrict__ gamma,
                const float* __restrict__ beta,
                float* __restrict__ out) {           // in: residual, out: final
    const int n   = blockIdx.x;
    const int tid = threadIdx.x;
    __shared__ int   sh_idx[128];
    __shared__ float red[128];

    const float4* hrow = (const float4*)(hs + (size_t)n * DIM_OUT);
    float4 acc = hrow[tid];                           // self loop contribution

    const int e0 = off[n], e1 = off[n + 1];
    for (int base = e0; base < e1; base += 128) {
        const int m = (e1 - base < 128) ? (e1 - base) : 128;
        if (tid < m) sh_idx[tid] = elist[base + tid];
        __syncthreads();
        int j = 0;
        for (; j + 4 <= m; j += 4) {                  // 4 loads in flight
            const float4 v0 = ((const float4*)(hs + (size_t)sh_idx[j + 0] * DIM_OUT))[tid];
            const float4 v1 = ((const float4*)(hs + (size_t)sh_idx[j + 1] * DIM_OUT))[tid];
            const float4 v2 = ((const float4*)(hs + (size_t)sh_idx[j + 2] * DIM_OUT))[tid];
            const float4 v3 = ((const float4*)(hs + (size_t)sh_idx[j + 3] * DIM_OUT))[tid];
            acc.x += (v0.x + v1.x) + (v2.x + v3.x);
            acc.y += (v0.y + v1.y) + (v2.y + v3.y);
            acc.z += (v0.z + v1.z) + (v2.z + v3.z);
            acc.w += (v0.w + v1.w) + (v2.w + v3.w);
        }
        for (; j < m; ++j) {
            const float4 v = ((const float4*)(hs + (size_t)sh_idx[j] * DIM_OUT))[tid];
            acc.x += v.x; acc.y += v.y; acc.z += v.z; acc.w += v.w;
        }
        __syncthreads();
    }

    const float nm = normv[n];
    const float4 r = ((const float4*)(out + (size_t)n * DIM_OUT))[tid];
    float4 y;
    y.x = acc.x * nm + r.x;  y.y = acc.y * nm + r.y;
    y.z = acc.z * nm + r.z;  y.w = acc.w * nm + r.w;

    // mean
    red[tid] = y.x + y.y + y.z + y.w;
    __syncthreads();
    #pragma unroll
    for (int s = 64; s > 0; s >>= 1) {
        if (tid < s) red[tid] += red[tid + s];
        __syncthreads();
    }
    const float mu = red[0] * (1.0f / (float)DIM_OUT);
    __syncthreads();
    // variance
    const float dx = y.x - mu, dy = y.y - mu, dz = y.z - mu, dw = y.w - mu;
    red[tid] = dx * dx + dy * dy + dz * dz + dw * dw;
    __syncthreads();
    #pragma unroll
    for (int s = 64; s > 0; s >>= 1) {
        if (tid < s) red[tid] += red[tid + s];
        __syncthreads();
    }
    const float inv = rsqrtf(red[0] * (1.0f / (float)DIM_OUT) + 1e-5f);

    const float4 g = ((const float4*)gamma)[tid];
    const float4 b = ((const float4*)beta)[tid];
    float4 o;
    o.x = fmaxf(dx * inv * g.x + b.x, 0.0f);
    o.y = fmaxf(dy * inv * g.y + b.y, 0.0f);
    o.z = fmaxf(dz * inv * g.z + b.z, 0.0f);
    o.w = fmaxf(dw * inv * g.w + b.w, 0.0f);
    ((float4*)(out + (size_t)n * DIM_OUT))[tid] = o;
}

// ---------------------------------------------------------------------------
extern "C" void kernel_launch(void* const* d_in, const int* in_sizes, int n_in,
                              void* d_out, int out_size, void* d_ws, size_t ws_size,
                              hipStream_t stream) {
    const float* x     = (const float*)d_in[0];
    const int*   src   = (const int*)  d_in[1];
    const int*   dst   = (const int*)  d_in[2];
    const float* W     = (const float*)d_in[3];
    const float* Wr    = (const float*)d_in[4];
    const float* br    = (const float*)d_in[5];
    const float* gamma = (const float*)d_in[6];
    const float* beta  = (const float*)d_in[7];
    float* out = (float*)d_out;

    const int nNodes = in_sizes[0] / DIM_IN;
    const int nEdges = in_sizes[1];

    // carve workspace (256B aligned regions)
    char* w = (char*)d_ws;
    auto carve = [&](size_t bytes) {
        void* p = (void*)w;
        w += (bytes + 255) & ~(size_t)255;
        return p;
    };
    int*          cnt    = (int*)         carve(sizeof(int) * nNodes);
    int*          cursor = (int*)         carve(sizeof(int) * nNodes);
    int*          off    = (int*)         carve(sizeof(int) * (nNodes + 1));
    float*        normv  = (float*)       carve(sizeof(float) * nNodes);
    int*          elist  = (int*)         carve(sizeof(int) * nEdges);
    unsigned int* xh     = (unsigned int*)carve(sizeof(unsigned int) * (size_t)nNodes * KPAIRS);
    unsigned int* Wp     = (unsigned int*)carve(sizeof(unsigned int) * KPAIRS * NPRIME);
    float*        hs     = (float*)       carve(sizeof(float) * (size_t)nNodes * DIM_OUT);
    (void)ws_size; (void)n_in; (void)out_size;

    // 1. zero counters
    zero_kernel<<<dim3((nNodes + 255) / 256), dim3(256), 0, stream>>>(cnt, nNodes);
    zero_kernel<<<dim3((nNodes + 255) / 256), dim3(256), 0, stream>>>(cursor, nNodes);
    // 2. degree count
    count_kernel<<<dim3((nEdges + 255) / 256), dim3(256), 0, stream>>>(dst, cnt, nEdges);
    // 3. scan -> offsets + norm
    scan_kernel<<<dim3(1), dim3(1024), 0, stream>>>(cnt, off, normv, nNodes);
    // 4. CSR fill
    fill_kernel<<<dim3((nEdges + 255) / 256), dim3(256), 0, stream>>>(
        src, dst, off, cursor, elist, nEdges);
    // 5. pack x -> f16
    {
        const int nPairs = nNodes * KPAIRS;
        convx_kernel<<<dim3((nPairs + 255) / 256), dim3(256), 0, stream>>>(x, xh, nPairs);
    }
    // 6. pack [W|Wr] -> WMMA B-fragment layout
    convw_kernel<<<dim3((KPAIRS * NPRIME + 255) / 256), dim3(256), 0, stream>>>(
        W, Wr, Wp, KPAIRS * NPRIME);
    // 7. fused WMMA GEMM: hs = (xW)*norm ; out = xWr + br
    {
        dim3 grid((nNodes + 63) / 64, NPRIME / 128);
        gemm_kernel<<<grid, dim3(256), 0, stream>>>(xh, Wp, normv, br, hs, out, nNodes);
    }
    // 8. aggregate + LN + ReLU
    agg_kernel<<<dim3(nNodes), dim3(128), 0, stream>>>(
        hs, off, elist, normv, gamma, beta, out);
}
